// GMMNet_82197084111198
// MI455X (gfx1250) — compile-verified, matrix operand order checked by reading, not run on hardware
//
#include <hip/hip_runtime.h>
#include <hip/hip_bf16.h>

#define NN 50000
#define EE 800000
#define HH 64
#define KK 4
#define GG 50
#define CC 10

typedef __attribute__((ext_vector_type(16))) _Float16 v16h;
typedef __attribute__((ext_vector_type(8)))  float    v8f;

// ---------------------------------------------------------------------------
// WMMA GEMM: C[M x Ncols] = A[M x 64] * B[64 x Ncols], f32 in/out, f16 compute.
// One wave (32 lanes) per 16x16 output tile; K=64 done as two K=32 WMMA steps.
// Fragment layouts per CDNA5 ISA 7.12.2 (16-bit A 16x32, B mirrored, f32 C/D).
// M and Ncols are multiples of 16 (N=50000=3125*16, cols in {64,256}).
// ---------------------------------------------------------------------------
__global__ void gemm64_wmma(const float* __restrict__ A,
                            const float* __restrict__ B,
                            float* __restrict__ C, int Ncols) {
    const int lane  = threadIdx.x & 31;
    const int mBase = blockIdx.x * 16;
    const int nBase = blockIdx.y * 16;
    const int row   = mBase + (lane & 15);
    const int col   = nBase + (lane & 15);
    const int khalf = (lane >> 4) ? 8 : 0;   // lanes 16-31 hold K+8 (A) / M+8 (C)

    v8f acc = {};
    #pragma unroll
    for (int kk = 0; kk < 64; kk += 32) {
        v16h a, b;
        #pragma unroll
        for (int v = 0; v < 8; ++v) {
            // ISA: VGPR 0-3 -> K = 2v,2v+1 (+8 for hi lanes); VGPR 4-7 -> K+16
            const int k = kk + khalf + ((v < 4) ? 2 * v : 16 + 2 * (v - 4));
            a[2 * v]     = (_Float16)A[(size_t)row * 64 + k];
            a[2 * v + 1] = (_Float16)A[(size_t)row * 64 + k + 1];
            b[2 * v]     = (_Float16)B[(size_t)k * Ncols + col];
            b[2 * v + 1] = (_Float16)B[(size_t)(k + 1) * Ncols + col];
        }
        acc = __builtin_amdgcn_wmma_f32_16x16x32_f16(
            /*neg_a=*/false, a, /*neg_b=*/false, b,
            /*c_mod=*/(short)0, acc, /*reuse_a=*/false, /*reuse_b=*/false);
    }
    // C/D layout: VGPR r -> M = r (lanes 0-15) / M = 8+r (lanes 16-31)
    const int mOut = mBase + khalf;
    #pragma unroll
    for (int r = 0; r < 8; ++r)
        C[(size_t)(mOut + r) * Ncols + col] = acc[r];
}

// ---------------------------------------------------------------------------
__global__ void zero_f32(float* __restrict__ p, long long n) {
    long long i = (long long)blockIdx.x * blockDim.x + threadIdx.x;
    if (i < n) p[i] = 0.0f;
}

__global__ void fill_u32(unsigned* __restrict__ p, unsigned v, int n) {
    int i = blockIdx.x * blockDim.x + threadIdx.x;
    if (i < n) p[i] = v;
}

__global__ void deg_kernel(const long long* __restrict__ ei,
                           float* __restrict__ deg) {
    int e = blockIdx.x * blockDim.x + threadIdx.x;
    if (e >= EE) return;
    int d = (int)ei[EE + e];
    unsafeAtomicAdd(&deg[d], 1.0f);
}

// Gaussian edge weights: gw[e,k] = exp(-0.5 * sum_d (ea-mu)^2 / (eps + sigma^2))
__global__ void gw_kernel(const float* __restrict__ ea,
                          const float* __restrict__ mu,
                          const float* __restrict__ sigma,
                          float* __restrict__ gw) {
    int e = blockIdx.x * blockDim.x + threadIdx.x;
    if (e >= EE) return;
    float e0 = ea[(size_t)e * 2], e1 = ea[(size_t)e * 2 + 1];
    #pragma unroll
    for (int k = 0; k < KK; ++k) {
        float d0 = e0 - mu[k * 2], d1 = e1 - mu[k * 2 + 1];
        float s0 = sigma[k * 2], s1 = sigma[k * 2 + 1];
        float q = d0 * d0 / (1e-15f + s0 * s0) + d1 * d1 / (1e-15f + s1 * s1);
        gw[(size_t)e * KK + k] = __expf(-0.5f * q);
    }
}

// msg[e,h] = sum_k gw[e,k] * xg[src,k*H+h]; agg[dst,h] += msg  (atomic)
// One thread per (edge, h). xg (51 MB) is L2-resident -> gather hits L2.
__global__ void edge_scatter(const long long* __restrict__ ei,
                             const float* __restrict__ gw,
                             const float* __restrict__ xg,
                             float* __restrict__ agg) {
    long long tid = (long long)blockIdx.x * blockDim.x + threadIdx.x;
    if (tid >= (long long)EE * HH) return;
    int e = (int)(tid >> 6);
    int h = (int)(tid & 63);
    int s = (int)ei[e];
    int d = (int)ei[EE + e];
    const float* xb = xg + (size_t)s * (KK * HH) + h;
    const float* w  = gw + (size_t)e * KK;
    float m = w[0] * xb[0] + w[1] * xb[HH] + w[2] * xb[2 * HH] + w[3] * xb[3 * HH];
    unsafeAtomicAdd(&agg[(size_t)d * HH + h], m);
}

// out = leaky_relu(agg/deg + xr + b)
__global__ void finalize_kernel(const float* __restrict__ agg,
                                const float* __restrict__ xr,
                                const float* __restrict__ deg,
                                const float* __restrict__ bias,
                                float* __restrict__ out) {
    long long tid = (long long)blockIdx.x * blockDim.x + threadIdx.x;
    if (tid >= (long long)NN * HH) return;
    int n = (int)(tid >> 6);
    int h = (int)(tid & 63);
    float dg = deg[n];
    dg = dg < 1.0f ? 1.0f : dg;
    float v = agg[tid] / dg + xr[tid] + bias[h];
    out[tid] = v > 0.0f ? v : 0.01f * v;
}

__global__ void add_kernel(const float* __restrict__ a,
                           const float* __restrict__ b,
                           float* __restrict__ o, long long n) {
    long long i = (long long)blockIdx.x * blockDim.x + threadIdx.x;
    if (i < n) o[i] = a[i] + b[i];
}

// segment_max over concat [x4, x1, x2, x3] using monotonic uint encoding
__device__ __forceinline__ unsigned enc_f32(float f) {
    unsigned u = __float_as_uint(f);
    return (u & 0x80000000u) ? ~u : (u | 0x80000000u);
}
__device__ __forceinline__ float dec_f32(unsigned e) {
    unsigned u = (e & 0x80000000u) ? (e & 0x7FFFFFFFu) : ~e;
    return __uint_as_float(u);
}

__global__ void pool_kernel(const float* __restrict__ x4,
                            const float* __restrict__ x1,
                            const float* __restrict__ x2,
                            const float* __restrict__ x3,
                            const long long* __restrict__ batch,
                            unsigned* __restrict__ pooled) {
    long long tid = (long long)blockIdx.x * blockDim.x + threadIdx.x;
    if (tid >= (long long)NN * 256) return;
    int node = (int)(tid >> 8);
    int c = (int)(tid & 255);
    const float* src = (c < 64) ? x4 : (c < 128) ? x1 : (c < 192) ? x2 : x3;
    float v = src[(size_t)node * HH + (c & 63)];
    int g = (int)batch[node];
    atomicMax(&pooled[g * 256 + c], enc_f32(v));
}

// head: z = relu(BN(pooled @ w1 + b1)); out = z @ w2 + b2  (G=50, tiny)
__global__ void head_kernel(const unsigned* __restrict__ pooled,
                            const float* __restrict__ w1,
                            const float* __restrict__ b1,
                            const float* __restrict__ gamma,
                            const float* __restrict__ beta,
                            const float* __restrict__ w2,
                            const float* __restrict__ b2,
                            float* __restrict__ out) {
    __shared__ float z[GG * HH];
    const int tid = threadIdx.x;
    const float bnscale = rsqrtf(1.0f + 1e-5f);
    for (int idx = tid; idx < GG * HH; idx += blockDim.x) {
        int g = idx >> 6, h = idx & 63;
        float s = b1[h];
        for (int j = 0; j < 256; ++j)
            s += dec_f32(pooled[g * 256 + j]) * w1[(size_t)j * HH + h];
        float zz = s * bnscale * gamma[h] + beta[h];
        z[idx] = zz > 0.0f ? zz : 0.0f;
    }
    __syncthreads();
    for (int idx = tid; idx < GG * CC; idx += blockDim.x) {
        int g = idx / CC, c = idx % CC;
        float s = b2[c];
        for (int h = 0; h < HH; ++h)
            s += z[g * HH + h] * w2[h * CC + c];
        out[idx] = s;
    }
}

// ---------------------------------------------------------------------------
extern "C" void kernel_launch(void* const* d_in, const int* in_sizes, int n_in,
                              void* d_out, int out_size, void* d_ws, size_t ws_size,
                              hipStream_t stream) {
    (void)in_sizes; (void)n_in; (void)out_size; (void)ws_size;

    const float*     x     = (const float*)d_in[0];
    const long long* ei    = (const long long*)d_in[1];   // int64 [2,E]
    const long long* batch = (const long long*)d_in[2];   // int64 [N]
    const float*     ea    = (const float*)d_in[3];

    const float *gp[4], *mup[4], *sgp[4], *rwp[4], *bp[4];
    for (int i = 0; i < 4; ++i) {
        gp[i]  = (const float*)d_in[4 + 5 * i + 0];
        mup[i] = (const float*)d_in[4 + 5 * i + 1];
        sgp[i] = (const float*)d_in[4 + 5 * i + 2];
        rwp[i] = (const float*)d_in[4 + 5 * i + 3];
        bp[i]  = (const float*)d_in[4 + 5 * i + 4];
    }
    const float* w1    = (const float*)d_in[24];
    const float* b1    = (const float*)d_in[25];
    const float* gamma = (const float*)d_in[26];
    const float* beta  = (const float*)d_in[27];
    const float* w2    = (const float*)d_in[28];
    const float* b2    = (const float*)d_in[29];

    // workspace carve (256B aligned)
    char* ws = (char*)d_ws;
    auto carve = [&](size_t bytes) -> void* {
        void* p = (void*)ws;
        ws += (bytes + 255) & ~(size_t)255;
        return p;
    };
    float*    xg     = (float*)carve((size_t)NN * KK * HH * 4);  // 51.2 MB
    float*    xr     = (float*)carve((size_t)NN * HH * 4);
    float*    agg    = (float*)carve((size_t)NN * HH * 4);
    float*    gw     = (float*)carve((size_t)EE * KK * 4);
    float*    deg    = (float*)carve((size_t)NN * 4);
    float*    x0     = (float*)carve((size_t)NN * HH * 4);
    float*    x1     = (float*)carve((size_t)NN * HH * 4);
    float*    x2     = (float*)carve((size_t)NN * HH * 4);
    float*    x3     = (float*)carve((size_t)NN * HH * 4);
    float*    x4     = (float*)carve((size_t)NN * HH * 4);
    unsigned* pooled = (unsigned*)carve((size_t)GG * 256 * 4);

    const long long nodeElems = (long long)NN * HH;
    const long long edgeElems = (long long)EE * HH;

    // degree (layer-invariant)
    zero_f32<<<(NN + 255) / 256, 256, 0, stream>>>(deg, NN);
    deg_kernel<<<(EE + 255) / 256, 256, 0, stream>>>(ei, deg);

    const float* lin[4]  = {x, x0, x1, x3};
    float*       lout[4] = {x0, x1, x2, x4};

    for (int i = 0; i < 4; ++i) {
        if (i == 3) {  // x3 = x0 + x2 (residual) feeds layer 3
            add_kernel<<<(int)((nodeElems + 255) / 256), 256, 0, stream>>>(
                x0, x2, x3, nodeElems);
        }
        // xg = lin @ g  [N x 256], xr = lin @ rw  [N x 64]  (WMMA)
        gemm64_wmma<<<dim3(NN / 16, (KK * HH) / 16), 32, 0, stream>>>(
            lin[i], gp[i], xg, KK * HH);
        gemm64_wmma<<<dim3(NN / 16, HH / 16), 32, 0, stream>>>(
            lin[i], rwp[i], xr, HH);
        zero_f32<<<(int)((nodeElems + 255) / 256), 256, 0, stream>>>(agg, nodeElems);
        gw_kernel<<<(EE + 255) / 256, 256, 0, stream>>>(ea, mup[i], sgp[i], gw);
        edge_scatter<<<(int)((edgeElems + 255) / 256), 256, 0, stream>>>(
            ei, gw, xg, agg);
        finalize_kernel<<<(int)((nodeElems + 255) / 256), 256, 0, stream>>>(
            agg, xr, deg, bp[i], lout[i]);
    }

    // pooling: init to encoded -FLT_MAX (bits 0xFF7FFFFF -> enc ~u = 0x00800000)
    fill_u32<<<(GG * 256 + 255) / 256, 256, 0, stream>>>(pooled, 0x00800000u, GG * 256);
    pool_kernel<<<(int)(((long long)NN * 256 + 255) / 256), 256, 0, stream>>>(
        x4, x1, x2, x3, batch, pooled);

    head_kernel<<<1, 256, 0, stream>>>(pooled, w1, b1, gamma, beta, w2, b2,
                                       (float*)d_out);
}
